// BaselineNeuralODE_40776419508625
// MI455X (gfx1250) — compile-verified, matrix operand order checked by reading, not run on hardware
//
#include <hip/hip_runtime.h>
#include <hip/hip_bf16.h>
#include <math.h>

// ---- problem sizes from the reference ----
#define TC_N 128
#define TT_N 256
#define F_N  512
#define L_N  256
#define EH_N 512   // ENC_H == DYN_H
#define G3_N 768   // 3*L

typedef _Float16 h8   __attribute__((ext_vector_type(8)));
typedef _Float16 v16h __attribute__((ext_vector_type(16)));
typedef float    v8f  __attribute__((ext_vector_type(8)));

__device__ __forceinline__ v16h cat8(h8 a, h8 b) {
  return __builtin_shufflevector(a, b, 0,1,2,3,4,5,6,7,8,9,10,11,12,13,14,15);
}

// Branchless fast tanh/sigmoid on the v_exp_f32/v_rcp_f32 hardware path.
__device__ __forceinline__ float fast_tanh(float x) {
  float ax = __builtin_fabsf(x);
  float e  = __expf(-2.0f * ax);                       // v_exp_f32
  float r  = (1.0f - e) * __builtin_amdgcn_rcpf(1.0f + e);
  return __builtin_copysignf(r, x);
}
__device__ __forceinline__ float fast_sigmoid(float x) {
  return __builtin_amdgcn_rcpf(1.0f + __expf(-x));
}

// One 16xN GEMM (K contraction) using v_wmma_f32_16x16x32_f16.
// A: 16 x K f16 row-major (LDS).  Bt: N x K f16 row-major (global, = W^T).
// 8 waves split N into TILES 16-wide tiles each.
// K-loop: unroll-1 over PAIRS of 32-wide K-chunks -> bounded live fragments
// (no spills, no cross-scan-step load hoisting), loads batched ahead of wmmas.
// MODE 0: out16[m*ldo+n] = f16(tanh(acc + bias[n]))
// MODE 1: out32[m*ldo+n] = acc + bias[n]
// MODE 2: RK4 final combine: ys[i] += dt*0.125*(k1 + 3*(k2+k3) + (acc+bias)),
//         with k1/k2/k3 at out32[i], out32[4096+i], out32[8192+i].
template<int TILES, int KCH, int MODE>
__device__ __forceinline__ void gemm16(
    const _Float16* __restrict__ A,
    const _Float16* __restrict__ Bt,
    const float*    __restrict__ bias,
    float*          __restrict__ out32,
    _Float16*       __restrict__ out16,
    int ldo, int wave, int lane,
    float dtc = 0.0f, float* __restrict__ ys = nullptr)
{
  constexpr int K = KCH * 32;
  static_assert((KCH & 1) == 0, "KCH must be even");
  const int m16  = lane & 15;
  const int half = lane >> 4;
  const _Float16* ap     = A  + m16 * K + half * 8;
  const _Float16* bpbase = Bt + (size_t)(wave * TILES * 16 + m16) * K + half * 8;

  v8f acc[TILES] = {};
  #pragma unroll 1
  for (int kc = 0; kc < KCH; kc += 2) {
    const int ko0 = kc * 32;
    const int ko1 = ko0 + 32;
    // batch-issue all loads for two K-steps (clause-friendly), then compute
    h8 a00 = *(const h8*)(ap + ko0);
    h8 a01 = *(const h8*)(ap + ko0 + 16);
    h8 a10 = *(const h8*)(ap + ko1);
    h8 a11 = *(const h8*)(ap + ko1 + 16);
    v16h b0[TILES], b1[TILES];
    #pragma unroll
    for (int t = 0; t < TILES; ++t) {
      const _Float16* bp = bpbase + (size_t)t * 16 * K;
      b0[t] = cat8(*(const h8*)(bp + ko0), *(const h8*)(bp + ko0 + 16));
      b1[t] = cat8(*(const h8*)(bp + ko1), *(const h8*)(bp + ko1 + 16));
    }
    v16h av0 = cat8(a00, a01);
    v16h av1 = cat8(a10, a11);
    #pragma unroll
    for (int t = 0; t < TILES; ++t)
      acc[t] = __builtin_amdgcn_wmma_f32_16x16x32_f16(
          false, av0, false, b0[t], (short)0, acc[t], false, false);
    #pragma unroll
    for (int t = 0; t < TILES; ++t)
      acc[t] = __builtin_amdgcn_wmma_f32_16x16x32_f16(
          false, av1, false, b1[t], (short)0, acc[t], false, false);
  }

  #pragma unroll
  for (int t = 0; t < TILES; ++t) {
    const int n = (wave * TILES + t) * 16 + m16;
    #pragma unroll
    for (int r = 0; r < 8; ++r) {
      const int m = half * 8 + r;
      const int i = m * ldo + n;
      float v = acc[t][r] + bias[n];
      if constexpr (MODE == 0) {
        out16[i] = (_Float16)fast_tanh(v);
      } else if constexpr (MODE == 1) {
        out32[i] = v;
      } else {
        float k1 = out32[i], k2 = out32[4096 + i], k3 = out32[8192 + i];
        ys[i] = ys[i] + dtc * 0.125f * (k1 + 3.0f * (k2 + k3) + v);
      }
    }
  }
}

// Transpose-and-convert: src (R x C) f32 row-major -> dst (C x R) f16 row-major.
__global__ void transpose_f16_kernel(const float* __restrict__ src,
                                     _Float16* __restrict__ dst, int R, int C) {
  int i = blockIdx.x * blockDim.x + threadIdx.x;
  if (i >= R * C) return;
  int c = i / R;          // dst row (N)
  int r = i - c * R;      // dst col (K)
  dst[i] = (_Float16)src[r * C + c];
}

__global__ __launch_bounds__(256)
void node_persistent_kernel(
    const float* __restrict__ ct,  const float* __restrict__ cv,
    const float* __restrict__ cm,  const float* __restrict__ tt,
    const float* __restrict__ enc_b1, const float* __restrict__ enc_b2,
    const float* __restrict__ gru_wi, const float* __restrict__ gru_bi,
    const float* __restrict__ gru_bh,
    const float* __restrict__ dyn_b1, const float* __restrict__ dyn_b2,
    const float* __restrict__ dec_b1, const float* __restrict__ dec_w2,
    const float* __restrict__ dec_b2,
    const _Float16* __restrict__ w1t_enc, const _Float16* __restrict__ w2t_enc,
    const _Float16* __restrict__ wht,
    const _Float16* __restrict__ w1t_dyn, const _Float16* __restrict__ w2t_dyn,
    const _Float16* __restrict__ wdect,
    float* __restrict__ out)
{
  __shared__ float    sH[16 * L_N];       // 16 KB  latent state (f32)
  __shared__ _Float16 sY16[16 * L_N];     //  8 KB  f16 GEMM-A staging
  __shared__ _Float16 sHid[16 * EH_N];    // 16 KB  f16 hidden activations
  __shared__ float    sK[16 * G3_N];      // 48 KB  k1/k2/k3 (4096 each) or gh(16x768)
  __shared__ float    sRed[256];
  __shared__ float    sEB1[EH_N], sEB2[L_N], sDB1[EH_N], sDB2[L_N];
  __shared__ float    sWi[G3_N], sBi[G3_N], sBh[G3_N];
  __shared__ float    sDecB1[L_N], sDecW2[L_N];
  __shared__ float    sObs[16], sMask[16];

  const int tid  = threadIdx.x;
  const int lane = tid & 31;
  const int wave = tid >> 5;
  const int f0   = blockIdx.x * 16;   // this block's 16 feature rows

  for (int i = tid; i < EH_N; i += 256) { sEB1[i] = enc_b1[i]; sDB1[i] = dyn_b1[i]; }
  for (int i = tid; i < L_N;  i += 256) {
    sEB2[i] = enc_b2[i]; sDB2[i] = dyn_b2[i];
    sDecB1[i] = dec_b1[i]; sDecW2[i] = dec_w2[i];
  }
  for (int i = tid; i < G3_N; i += 256) { sWi[i] = gru_wi[i]; sBi[i] = gru_bi[i]; sBh[i] = gru_bh[i]; }
  for (int i = tid; i < 16 * L_N; i += 256) sH[i] = 0.0f;
  __syncthreads();

  // 3/8-rule RK4 step of f(y)=tanh(y@W1+b1)@W2+b2, state in sH.
  auto rk4 = [&](float dt, const _Float16* w1t, const _Float16* w2t,
                 const float* b1, const float* b2) {
    const float third = 1.0f / 3.0f;
    for (int i = tid; i < 16 * L_N; i += 256) sY16[i] = (_Float16)sH[i];
    __syncthreads();
    #pragma unroll 1
    for (int e = 0; e < 3; ++e) {
      gemm16<4, L_N / 32, 0>(sY16, w1t, b1, nullptr, sHid, EH_N, wave, lane);
      __syncthreads();
      gemm16<2, EH_N / 32, 1>(sHid, w2t, b2, sK + e * 4096, nullptr, L_N, wave, lane);
      __syncthreads();
      for (int i = tid; i < 16 * L_N; i += 256) {
        float y = sH[i];
        float v;
        if (e == 0)      v = y + dt * third * sK[i];
        else if (e == 1) v = y + dt * (sK[4096 + i] - third * sK[i]);
        else             v = y + dt * (sK[i] - sK[4096 + i] + sK[8192 + i]);
        sY16[i] = (_Float16)v;
      }
      __syncthreads();
    }
    // k4 eval fused with the final RK4 combine (writes sH in the epilogue)
    gemm16<4, L_N / 32, 0>(sY16, w1t, b1, nullptr, sHid, EH_N, wave, lane);
    __syncthreads();
    gemm16<2, EH_N / 32, 2>(sHid, w2t, b2, sK, nullptr, L_N, wave, lane, dt, sH);
    __syncthreads();
  };

  // GRU observation update at context index idx.
  auto gru = [&](int idx) {
    if (tid < 16) { sObs[tid] = cv[idx * F_N + f0 + tid]; sMask[tid] = cm[idx * F_N + f0 + tid]; }
    for (int i = tid; i < 16 * L_N; i += 256) sY16[i] = (_Float16)sH[i];
    __syncthreads();
    gemm16<6, L_N / 32, 1>(sY16, wht, sBh, sK, nullptr, G3_N, wave, lane);  // gh = h@Wh+bh
    __syncthreads();
    for (int i = tid; i < 16 * L_N; i += 256) {
      int m = i >> 8, c = i & 255;
      float ob = sObs[m], mk = sMask[m], h = sH[i];
      float ghr = sK[m * G3_N + c];
      float ghz = sK[m * G3_N + 256 + c];
      float ghn = sK[m * G3_N + 512 + c];
      float gir = ob * sWi[c]       + sBi[c];
      float giz = ob * sWi[256 + c] + sBi[256 + c];
      float gin = ob * sWi[512 + c] + sBi[512 + c];
      float r  = fast_sigmoid(gir + ghr);
      float zg = fast_sigmoid(giz + ghz);
      float n  = fast_tanh(gin + r * ghn);
      float hc = (1.0f - zg) * n + zg * h;
      sH[i] = mk * hc + (1.0f - mk) * h;
    }
    __syncthreads();
  };

  // Decoder: preds[s, f] = tanh(z@Wd1+bd1)@Wd2 + bd2
  auto decode = [&](int s) {
    for (int i = tid; i < 16 * L_N; i += 256) sY16[i] = (_Float16)sH[i];
    __syncthreads();
    gemm16<2, L_N / 32, 0>(sY16, wdect, sDecB1, nullptr, sHid, L_N, wave, lane);
    __syncthreads();
    {
      int m = tid >> 4, seg = tid & 15;
      float p = 0.0f;
      #pragma unroll
      for (int j = 0; j < 16; ++j) {
        int c = seg * 16 + j;
        p += (float)sHid[m * L_N + c] * sDecW2[c];
      }
      sRed[tid] = p;
    }
    __syncthreads();
    if (tid < 16) {
      float sum = dec_b2[0];
      #pragma unroll
      for (int j = 0; j < 16; ++j) sum += sRed[tid * 16 + j];
      out[s * F_N + f0 + tid] = sum;
    }
    __syncthreads();
  };

  // ---- encoder: reverse scan over context ----
  #pragma unroll 1
  for (int s = 0; s < TC_N; ++s) {
    int idx = TC_N - 1 - s;
    float dt = (s == 0) ? 0.0f : (ct[idx + 1] - ct[idx]);   // uniform across block
    if (dt > 0.0f) rk4(dt, w1t_enc, w2t_enc, sEB1, sEB2);
    gru(idx);
  }

  // ---- latent scan + immediate decode ----
  decode(0);
  #pragma unroll 1
  for (int s = 1; s < TT_N; ++s) {
    float dt = tt[s] - tt[s - 1];
    rk4(dt, w1t_dyn, w2t_dyn, sDB1, sDB2);
    decode(s);
  }
}

extern "C" void kernel_launch(void* const* d_in, const int* in_sizes, int n_in,
                              void* d_out, int out_size, void* d_ws, size_t ws_size,
                              hipStream_t stream) {
  const float* ct     = (const float*)d_in[0];
  const float* cv     = (const float*)d_in[1];
  const float* cm     = (const float*)d_in[2];
  const float* tt     = (const float*)d_in[3];
  const float* enc_w1 = (const float*)d_in[4];
  const float* enc_b1 = (const float*)d_in[5];
  const float* enc_w2 = (const float*)d_in[6];
  const float* enc_b2 = (const float*)d_in[7];
  const float* gru_wi = (const float*)d_in[8];
  const float* gru_bi = (const float*)d_in[9];
  const float* gru_wh = (const float*)d_in[10];
  const float* gru_bh = (const float*)d_in[11];
  const float* dyn_w1 = (const float*)d_in[12];
  const float* dyn_b1 = (const float*)d_in[13];
  const float* dyn_w2 = (const float*)d_in[14];
  const float* dyn_b2 = (const float*)d_in[15];
  const float* dec_w1 = (const float*)d_in[16];
  const float* dec_b1 = (const float*)d_in[17];
  const float* dec_w2 = (const float*)d_in[18];
  const float* dec_b2 = (const float*)d_in[19];

  // f16 transposed weight staging in workspace (N x K row-major each)
  _Float16* w1te = (_Float16*)d_ws;                // 512*256
  _Float16* w2te = w1te + 512 * 256;               // 256*512
  _Float16* whT  = w2te + 256 * 512;               // 768*256
  _Float16* w1td = whT  + 768 * 256;               // 512*256
  _Float16* w2td = w1td + 512 * 256;               // 256*512
  _Float16* wdec = w2td + 256 * 512;               // 256*256

  auto tp = [&](const float* src, _Float16* dst, int R, int C) {
    int total = R * C;
    transpose_f16_kernel<<<(total + 255) / 256, 256, 0, stream>>>(src, dst, R, C);
  };
  tp(enc_w1, w1te, 256, 512);
  tp(enc_w2, w2te, 512, 256);
  tp(gru_wh, whT,  256, 768);
  tp(dyn_w1, w1td, 256, 512);
  tp(dyn_w2, w2td, 512, 256);
  tp(dec_w1, wdec, 256, 256);

  node_persistent_kernel<<<F_N / 16, 256, 0, stream>>>(
      ct, cv, cm, tt,
      enc_b1, enc_b2, gru_wi, gru_bi, gru_bh,
      dyn_b1, dyn_b2, dec_b1, dec_w2, dec_b2,
      w1te, w2te, whT, w1td, w2td, wdec,
      (float*)d_out);
}
